// CustomGRU_77197742178494
// MI455X (gfx1250) — matrix operand order repaired and checked by reading, not compiled, not run on hardware
//
#include <hip/hip_runtime.h>
#include <hip/hip_bf16.h>

// GRU: B=64, S=512, I=512, H=1024 (fp32 in/out), bf16 WMMA compute.
// Phase 0 (parallel): proj_g[t][b][h] = x_t @ Wg^T + bg for g in {z,r,h}.
// Recurrence: ONE persistent kernel, 128 blocks looping over all 512 steps
// with a device-scope software barrier between the z/r phase and the h phase.

static constexpr int GB = 64;     // batch
static constexpr int GS = 512;    // seq len
static constexpr int GI = 512;    // input dim
static constexpr int GH = 1024;   // hidden dim
static constexpr unsigned NBLK = 128;  // persistent grid size

typedef __attribute__((ext_vector_type(16))) __bf16 v16bf;
typedef __attribute__((ext_vector_type(8)))  __bf16 v8bf;
typedef __attribute__((ext_vector_type(8)))  float  v8f;

__device__ __forceinline__ __bf16 f2bf(float f) {
  union { float f; unsigned u; } x; x.f = f;
  unsigned r = x.u + 0x7FFFu + ((x.u >> 16) & 1u);  // round-to-nearest-even
  union { unsigned short s; __bf16 b; } y;
  y.s = (unsigned short)(r >> 16);
  return y.b;
}

// A-operand tile (16x32 bf16) from a bf16 row-major source.
// lanes 0-15: M=lane,    K = {kk+0..15}
// lanes 16-31: M=lane-16, K = {kk+16..31}
__device__ __forceinline__ v16bf load_a_tile(const __bf16* __restrict__ rowptr,
                                             int kk, int lane) {
  const int ks = lane & 16;  // 0 or 16
  const v8bf lo = *(const v8bf*)(rowptr + kk + ks);
  const v8bf hi = *(const v8bf*)(rowptr + kk + ks + 8);
  return __builtin_shufflevector(lo, hi, 0, 1, 2, 3, 4, 5, 6, 7,
                                 8, 9, 10, 11, 12, 13, 14, 15);
}

__global__ void f32_to_bf16_k(const float* __restrict__ s,
                              __bf16* __restrict__ d, int n) {
  int i = blockIdx.x * blockDim.x + threadIdx.x;
  if (i < n) d[i] = f2bf(s[i]);
}

__global__ void init_bar_k(unsigned* p) { p[0] = 0u; p[1] = 0u; }

// Device-wide barrier for a persistent grid of NBLK co-resident blocks.
// Ticket counter + generation flag, agent-scope acquire/release.
__device__ __forceinline__ void grid_sync(unsigned* __restrict__ cnt,
                                          unsigned* __restrict__ gen) {
  __threadfence();     // make this thread's writes agent-visible
  __syncthreads();     // all block writes precede thread 0's arrival
  if (threadIdx.x == 0) {
    const unsigned g =
        __hip_atomic_load(gen, __ATOMIC_RELAXED, __HIP_MEMORY_SCOPE_AGENT);
    const unsigned t = __hip_atomic_fetch_add(cnt, 1u, __ATOMIC_ACQ_REL,
                                              __HIP_MEMORY_SCOPE_AGENT);
    if (t == NBLK - 1u) {
      __hip_atomic_store(cnt, 0u, __ATOMIC_RELAXED, __HIP_MEMORY_SCOPE_AGENT);
      __hip_atomic_store(gen, g + 1u, __ATOMIC_RELEASE,
                         __HIP_MEMORY_SCOPE_AGENT);
    } else {
      while (__hip_atomic_load(gen, __ATOMIC_ACQUIRE,
                               __HIP_MEMORY_SCOPE_AGENT) == g)
        __builtin_amdgcn_s_sleep(1);
    }
  }
  __syncthreads();
  __threadfence();     // acquire: see other blocks' writes
}

// ---------------------------------------------------------------------------
// Phase 0: big parallel GEMM. proj_g[t][b][h] = x[b][t][:] . Wg[h][:] + bg[h].
// One wave per 16(b) x 64(h) tile at one (gate, t). 98304 waves total.
// ---------------------------------------------------------------------------
__global__ __launch_bounds__(128) void gru_xproj(
    const __bf16* __restrict__ x_bf,
    const __bf16* __restrict__ Wz, const __bf16* __restrict__ Wr,
    const __bf16* __restrict__ Wh,
    const float* __restrict__ bz, const float* __restrict__ br,
    const float* __restrict__ bh, float* __restrict__ proj) {
  const int lane = threadIdx.x & 31;
  const int wave = threadIdx.x >> 5;
  int gid = blockIdx.x * 4 + wave;
  const int s = gid & (GS - 1);  gid >>= 9;
  const int bt = gid & 3;        gid >>= 2;
  const int strip = gid & 15;    gid >>= 4;
  const int gate = gid;          // 0..2

  const __bf16* Wg = (gate == 0) ? Wz : (gate == 1) ? Wr : Wh;
  const float*  bg = (gate == 0) ? bz : (gate == 1) ? br : bh;
  float* pg = proj + (size_t)gate * GS * GB * GH;

  const int nb   = strip * 64;
  const int mrow = bt * 16 + (lane & 15);   // batch index
  const int colL = nb + (lane & 15);
  const int bofs = lane & 16;

  v8f acc[4];
#pragma unroll
  for (int q = 0; q < 4; ++q) {
    const float bv = bg[colL + q * 16];
#pragma unroll
    for (int r = 0; r < 8; ++r) acc[q][r] = bv;
  }

  const __bf16* xrow = x_bf + (size_t)mrow * GS * GI + (size_t)s * GI;
  for (int kk = 0; kk < GI; kk += 32) {
    const v16bf a = load_a_tile(xrow, kk, lane);
#pragma unroll
    for (int q = 0; q < 4; ++q) {
      const v16bf b = *(const v16bf*)(Wg + (size_t)(colL + q * 16) * GI + kk + bofs);
      acc[q] = __builtin_amdgcn_wmma_f32_16x16x32_bf16(
          false, a, false, b, (short)0, acc[q], false, false);
    }
  }

  const int rbase = bt * 16 + ((lane & 16) >> 1);
#pragma unroll
  for (int q = 0; q < 4; ++q) {
    const int col = colL + q * 16;
#pragma unroll
    for (int r = 0; r < 8; ++r) {
      const int row = rbase + r;
      pg[((size_t)s * GB + row) * GH + col] = acc[q][r];
    }
  }
}

// ---------------------------------------------------------------------------
// Recurrent core helper: 16x64 tile of (A @ Ug^T), K=GH split over the
// block's 4 waves (256 each, double-buffered), reduced through LDS.
// Result valid in wave 0's acc only.
// ---------------------------------------------------------------------------
__device__ __forceinline__ void ktile_gemm(
    const __bf16* __restrict__ arow, const __bf16* __restrict__ Ug,
    int colL, int bofs, int lane, int wave,
    float (*red)[32][32], v8f acc[4]) {
#pragma unroll
  for (int q = 0; q < 4; ++q)
#pragma unroll
    for (int r = 0; r < 8; ++r) acc[q][r] = 0.0f;

  const int kbeg = wave * (GH / 4);
  const int kend = kbeg + (GH / 4);

  v16bf a_cur = load_a_tile(arow, kbeg, lane);
  v16bf b_cur[4], b_nxt[4];
#pragma unroll
  for (int q = 0; q < 4; ++q)
    b_cur[q] = *(const v16bf*)(Ug + (size_t)(colL + q * 16) * GH + kbeg + bofs);

  for (int kk = kbeg; kk < kend; kk += 32) {
    const int kn = (kk + 32 < kend) ? kk + 32 : kbeg;  // dummy reload on last
    const v16bf a_nxt = load_a_tile(arow, kn, lane);
#pragma unroll
    for (int q = 0; q < 4; ++q)
      b_nxt[q] = *(const v16bf*)(Ug + (size_t)(colL + q * 16) * GH + kn + bofs);
#pragma unroll
    for (int q = 0; q < 4; ++q)
      acc[q] = __builtin_amdgcn_wmma_f32_16x16x32_bf16(
          false, a_cur, false, b_cur[q], (short)0, acc[q], false, false);
    a_cur = a_nxt;
#pragma unroll
    for (int q = 0; q < 4; ++q) b_cur[q] = b_nxt[q];
  }

  // Cross-wave reduction via LDS (fixed order -> deterministic).
#pragma unroll
  for (int q = 0; q < 4; ++q)
    *(v8f*)&red[wave][lane][q * 8] = acc[q];
  __syncthreads();
  if (wave == 0) {
#pragma unroll
    for (int q = 0; q < 4; ++q) {
      const v8f p1 = *(const v8f*)&red[1][lane][q * 8];
      const v8f p2 = *(const v8f*)&red[2][lane][q * 8];
      const v8f p3 = *(const v8f*)&red[3][lane][q * 8];
#pragma unroll
      for (int r = 0; r < 8; ++r) acc[q][r] += p1[r] + p2[r] + p3[r];
    }
  }
}

// ---------------------------------------------------------------------------
// Persistent recurrence kernel: 128 co-resident blocks, t = 0..S-1 in-kernel.
// Per step: phase 1 (z & r gates, all 128 blocks), grid barrier,
//           phase 2 (h update, blocks 0..63), grid barrier.
// ---------------------------------------------------------------------------
__global__ __launch_bounds__(128) void gru_recurrent(
    const __bf16* __restrict__ Uz, const __bf16* __restrict__ Ur,
    const __bf16* __restrict__ Uh,
    const float* __restrict__ proj_z, const float* __restrict__ proj_r,
    const float* __restrict__ proj_h,
    float* __restrict__ h_buf, __bf16* __restrict__ h_bf,
    float* __restrict__ z_buf, __bf16* __restrict__ rh_bf,
    float* __restrict__ out, unsigned* __restrict__ bar) {
  __shared__ float red[4][32][32];
  unsigned* cnt = bar;
  unsigned* gen = bar + 1;

  const int lane = threadIdx.x & 31;
  const int wave = threadIdx.x >> 5;
  const int bofs = lane & 16;

  // Phase-1 tile: (mt 0..3, strip 0..31); strips >= 16 compute the r gate.
  const int mt1    = blockIdx.x & 3;
  const int strip1 = blockIdx.x >> 2;
  const int gate_r = (strip1 >= 16);
  const int colL1  = (strip1 & 15) * 64 + (lane & 15);
  const int mrow1  = mt1 * 16 + (lane & 15);
  const int rbase1 = mt1 * 16 + ((lane & 16) >> 1);
  const __bf16* Ug1 = gate_r ? Ur : Uz;

  // Phase-2 tile (blocks 0..63): (mt 0..3, strip 0..15).
  const bool p2 = (blockIdx.x < 64);
  const int mt2    = blockIdx.x & 3;
  const int strip2 = (blockIdx.x >> 2) & 15;
  const int colL2  = strip2 * 64 + (lane & 15);
  const int mrow2  = mt2 * 16 + (lane & 15);
  const int rbase2 = mt2 * 16 + ((lane & 16) >> 1);

  for (int t = 0; t < GS; ++t) {
    // ---- Phase 1: z = sigmoid(proj_z + h Uz^T); rh = sigmoid(proj_r + h Ur^T)*h
    {
      v8f acc[4];
      ktile_gemm(h_bf + (size_t)mrow1 * GH, Ug1, colL1, bofs, lane, wave,
                 red, acc);
      if (wave == 0) {
        const float* pg = (gate_r ? proj_r : proj_z) + (size_t)t * GB * GH;
#pragma unroll
        for (int q = 0; q < 4; ++q) {
          const int col = colL1 + q * 16;
#pragma unroll
          for (int r = 0; r < 8; ++r) {
            const int row = rbase1 + r;
            const float v = acc[q][r] + pg[(size_t)row * GH + col];
            const float g = 1.0f / (1.0f + __expf(-v));
            if (gate_r) rh_bf[row * GH + col] = f2bf(g * h_buf[row * GH + col]);
            else        z_buf[row * GH + col] = g;
          }
        }
      }
    }
    grid_sync(cnt, gen);

    // ---- Phase 2: hhat = tanh(proj_h + rh Uh^T); h = (1-z)h + z*hhat
    if (p2) {
      v8f acc[4];
      ktile_gemm(rh_bf + (size_t)mrow2 * GH, Uh, colL2, bofs, lane, wave,
                 red, acc);
      if (wave == 0) {
        const float* pg = proj_h + (size_t)t * GB * GH;
#pragma unroll
        for (int q = 0; q < 4; ++q) {
          const int col = colL2 + q * 16;
#pragma unroll
          for (int r = 0; r < 8; ++r) {
            const int row = rbase2 + r;
            const float hh = tanhf(acc[q][r] + pg[(size_t)row * GH + col]);
            const float z  = z_buf[row * GH + col];
            const float hp = h_buf[row * GH + col];
            const float hn = (1.0f - z) * hp + z * hh;
            h_buf[row * GH + col] = hn;
            h_bf [row * GH + col] = f2bf(hn);
            out[(size_t)row * GS * GH + (size_t)t * GH + col] = hn;
            if (t == GS - 1)
              out[(size_t)GB * GS * GH + (size_t)row * GH + col] = hn;
          }
        }
      }
    }
    grid_sync(cnt, gen);
  }
}

extern "C" void kernel_launch(void* const* d_in, const int* in_sizes, int n_in,
                              void* d_out, int out_size, void* d_ws, size_t ws_size,
                              hipStream_t stream) {
  (void)in_sizes; (void)n_in; (void)out_size; (void)ws_size;
  const float* x  = (const float*)d_in[0];
  const float* h0 = (const float*)d_in[1];
  const float* Wz = (const float*)d_in[2];
  const float* bz = (const float*)d_in[3];
  const float* Uz = (const float*)d_in[4];
  const float* Wr = (const float*)d_in[5];
  const float* br = (const float*)d_in[6];
  const float* Ur = (const float*)d_in[7];
  const float* Wh = (const float*)d_in[8];
  const float* bh = (const float*)d_in[9];
  const float* Uh = (const float*)d_in[10];
  float* out = (float*)d_out;

  const size_t nX = (size_t)GB * GS * GI;  // 16,777,216
  const size_t nW = (size_t)GH * GI;       // 524,288
  const size_t nU = (size_t)GH * GH;       // 1,048,576
  const size_t nH = (size_t)GB * GH;       // 65,536
  const size_t nP = (size_t)GS * GB * GH;  // 33,554,432 per gate

  __bf16* x_bf = (__bf16*)d_ws;
  __bf16* Wz_b = x_bf + nX;
  __bf16* Wr_b = Wz_b + nW;
  __bf16* Wh_b = Wr_b + nW;
  __bf16* Uz_b = Wh_b + nW;
  __bf16* Ur_b = Uz_b + nU;
  __bf16* Uh_b = Ur_b + nU;
  __bf16* h_bf  = Uh_b + nU;
  __bf16* rh_bf = h_bf + nH;
  float*  h_buf = (float*)(rh_bf + nH);
  float*  z_buf = h_buf + nH;
  float*  proj  = z_buf + nH;              // 3 gates x [S][B][H] f32
  float*  proj_z = proj;
  float*  proj_r = proj + nP;
  float*  proj_h = proj + 2 * nP;
  unsigned* bar  = (unsigned*)(proj + 3 * nP);  // {count, generation}

  // Conversions (deterministic, redone each call).
  const int cb = 256;
  f32_to_bf16_k<<<(int)((nX + cb - 1) / cb), cb, 0, stream>>>(x, x_bf, (int)nX);
  f32_to_bf16_k<<<(int)((nW + cb - 1) / cb), cb, 0, stream>>>(Wz, Wz_b, (int)nW);
  f32_to_bf16_k<<<(int)((nW + cb - 1) / cb), cb, 0, stream>>>(Wr, Wr_b, (int)nW);
  f32_to_bf16_k<<<(int)((nW + cb - 1) / cb), cb, 0, stream>>>(Wh, Wh_b, (int)nW);
  f32_to_bf16_k<<<(int)((nU + cb - 1) / cb), cb, 0, stream>>>(Uz, Uz_b, (int)nU);
  f32_to_bf16_k<<<(int)((nU + cb - 1) / cb), cb, 0, stream>>>(Ur, Ur_b, (int)nU);
  f32_to_bf16_k<<<(int)((nU + cb - 1) / cb), cb, 0, stream>>>(Uh, Uh_b, (int)nU);
  f32_to_bf16_k<<<(int)((nH + cb - 1) / cb), cb, 0, stream>>>(h0, h_bf, (int)nH);
  hipMemcpyAsync(h_buf, h0, nH * sizeof(float), hipMemcpyDeviceToDevice, stream);
  init_bar_k<<<1, 1, 0, stream>>>(bar);

  // Phase 0: all input projections in one big WMMA GEMM (high occupancy).
  gru_xproj<<<24576, 128, 0, stream>>>(x_bf, Wz_b, Wr_b, Wh_b, bz, br, bh, proj);

  // Whole 512-step recurrence in one persistent kernel.
  gru_recurrent<<<NBLK, 128, 0, stream>>>(Uz_b, Ur_b, Uh_b, proj_z, proj_r,
                                          proj_h, h_buf, h_bf, z_buf, rh_bf,
                                          out, bar);
}
